// TopoSAModule_4758823764634
// MI455X (gfx1250) — matrix operand order repaired and checked by reading, not compiled, not run on hardware
//
#include <hip/hip_runtime.h>
#include <hip/hip_bf16.h>

typedef __attribute__((ext_vector_type(16))) __bf16 v16bf;
typedef __attribute__((ext_vector_type(8)))  float  v8f;

#define CDIM   128
#define LDA    136      // padded LDS row stride in ushorts (272B -> conflict-free)
#define KGROUP 8
#define ROWS   128      // rows per macro-tile (16 points * 8 neighbors, or 128 points)

__device__ __forceinline__ unsigned short f2bf(float f) {
  unsigned int u = __float_as_uint(f);
  u += 0x7FFFu + ((u >> 16) & 1u);      // round-to-nearest-even
  return (unsigned short)(u >> 16);
}

union Frag16 { v16bf bf; uint4 q[2]; };

// A fragment (16x32 bf16): lane m=l&15; half-lanes select K+8.
// Per lane: halfwords [kb, kb+8) and [kb+16, kb+24).
__device__ __forceinline__ v16bf load_fragA(const unsigned short* base, int row0, int kk, int lane) {
  int m  = row0 + (lane & 15);
  int kb = kk + ((lane >> 4) << 3);
  const unsigned short* r = base + m * LDA;
  Frag16 f;
  f.q[0] = *reinterpret_cast<const uint4*>(r + kb);
  f.q[1] = *reinterpret_cast<const uint4*>(r + kb + 16);
  return f.bf;
}

// B fragment (32x16 bf16) from transposed weight WT[n][k]:
// lane n=l&15; half-lanes select K+8; halfwords [kb, kb+8) and [kb+16, kb+24).
__device__ __forceinline__ v16bf load_fragB(const unsigned short* baseT, int n0, int kk, int lane) {
  int n  = n0 + (lane & 15);
  int kb = kk + ((lane >> 4) << 3);
  const unsigned short* r = baseT + n * LDA;
  Frag16 f;
  f.q[0] = *reinterpret_cast<const uint4*>(r + kb);
  f.q[1] = *reinterpret_cast<const uint4*>(r + kb + 16);
  return f.bf;
}

// Stage W (row-major [k][n] f32) into LDS as bf16 transposed: sWT[n][k].
__device__ __forceinline__ void load_weightT(const float* __restrict__ W,
                                             unsigned short* sWT, int tid) {
  for (int i = tid; i < CDIM * CDIM; i += 256) {
    int k = i >> 7, n = i & 127;
    sWT[n * LDA + k] = f2bf(W[i]);
  }
}

// ---------------- Stage 1: grouped PointNet unit ----------------
__global__ __launch_bounds__(256)
void pointnet_kernel(const float* __restrict__ x,
                     const float* __restrict__ W1, const float* __restrict__ b1,
                     const float* __restrict__ W2, const float* __restrict__ b2,
                     const int* __restrict__ sidx, const int* __restrict__ gidx,
                     float* __restrict__ pf, int M) {
  __shared__ unsigned short sW1[CDIM * LDA];
  __shared__ unsigned short sW2[CDIM * LDA];
  __shared__ unsigned short sA [ROWS * LDA];
  __shared__ unsigned short sH [ROWS * LDA];

  const int tid  = threadIdx.x;
  const int wave = tid >> 5;
  const int lane = tid & 31;

  load_weightT(W1, sW1, tid);
  load_weightT(W2, sW2, tid);
  __syncthreads();

  const int col = wave * 16 + (lane & 15);
  const float bias1 = b1[col];
  const float bias2 = b2[col];

  const int ntiles = M / 16;                      // 16 points per tile
  for (int tile = blockIdx.x; tile < ntiles; tile += gridDim.x) {
    const int pt0 = tile * 16;

    // Gather 128 rows (double indirection), convert f32->bf16 into LDS.
    // wave w owns rows [16w, 16w+16); 32 lanes x float4 cover one 128-f32 row.
    for (int i = 0; i < 16; ++i) {
      int r = wave * 16 + i;
      int p = pt0 + (r >> 3);
      int g = gidx[p * KGROUP + (r & 7)];
      long long src = sidx[g];
      float4 v = reinterpret_cast<const float4*>(x + src * CDIM)[lane];
      unsigned short* d = sA + r * LDA + lane * 4;
      d[0] = f2bf(v.x); d[1] = f2bf(v.y); d[2] = f2bf(v.z); d[3] = f2bf(v.w);
    }
    __syncthreads();

    // GEMM1: h = relu(A @ W1 + b1) -> sH (bf16)
    for (int t = 0; t < 8; ++t) {
      v8f acc = {};
      for (int kk = 0; kk < CDIM; kk += 32) {
        v16bf a = load_fragA(sA,  t * 16,    kk, lane);
        v16bf b = load_fragB(sW1, wave * 16, kk, lane);
        acc = __builtin_amdgcn_wmma_f32_16x16x32_bf16(false, a, false, b,
                                                      (short)0, acc, false, false);
      }
      int rbase = t * 16 + ((lane >> 4) << 3);    // D layout: M = g + 8*(lane>=16)
      for (int g = 0; g < 8; ++g) {
        float v = acc[g] + bias1;
        v = v > 0.0f ? v : 0.0f;
        sH[(rbase + g) * LDA + col] = f2bf(v);
      }
    }
    __syncthreads();

    // GEMM2 + max over the 8 neighbor rows (they are exactly the 8 acc VGPRs)
    for (int t = 0; t < 8; ++t) {
      v8f acc = {};
      for (int kk = 0; kk < CDIM; kk += 32) {
        v16bf a = load_fragA(sH,  t * 16,    kk, lane);
        v16bf b = load_fragB(sW2, wave * 16, kk, lane);
        acc = __builtin_amdgcn_wmma_f32_16x16x32_bf16(false, a, false, b,
                                                      (short)0, acc, false, false);
      }
      float mx = acc[0];
      #pragma unroll
      for (int g = 1; g < 8; ++g) mx = fmaxf(mx, acc[g]);
      int p = pt0 + t * 2 + (lane >> 4);
      pf[(long long)p * CDIM + col] = mx + bias2;
    }
    __syncthreads();
  }
}

// ---------------- Stage 2: xw = point_feat @ Wg ----------------
__global__ __launch_bounds__(256)
void gcn_gemm_kernel(const float* __restrict__ pf, const float* __restrict__ Wg,
                     float* __restrict__ xw, int M) {
  __shared__ unsigned short sW[CDIM * LDA];
  __shared__ unsigned short sA[ROWS * LDA];
  const int tid  = threadIdx.x;
  const int wave = tid >> 5;
  const int lane = tid & 31;

  load_weightT(Wg, sW, tid);
  __syncthreads();

  const int col = wave * 16 + (lane & 15);
  const int ntiles = (M + ROWS - 1) / ROWS;
  for (int tile = blockIdx.x; tile < ntiles; tile += gridDim.x) {
    const int p0 = tile * ROWS;
    for (int i = 0; i < 16; ++i) {
      int r = wave * 16 + i;
      int p = p0 + r;
      unsigned short* d = sA + r * LDA + lane * 4;
      if (p < M) {
        float4 v = reinterpret_cast<const float4*>(pf + (long long)p * CDIM)[lane];
        d[0] = f2bf(v.x); d[1] = f2bf(v.y); d[2] = f2bf(v.z); d[3] = f2bf(v.w);
      } else {
        d[0] = 0; d[1] = 0; d[2] = 0; d[3] = 0;
      }
    }
    __syncthreads();
    for (int t = 0; t < 8; ++t) {
      v8f acc = {};
      for (int kk = 0; kk < CDIM; kk += 32) {
        v16bf a = load_fragA(sA, t * 16,    kk, lane);
        v16bf b = load_fragB(sW, wave * 16, kk, lane);
        acc = __builtin_amdgcn_wmma_f32_16x16x32_bf16(false, a, false, b,
                                                      (short)0, acc, false, false);
      }
      int rbase = t * 16 + ((lane >> 4) << 3);
      #pragma unroll
      for (int g = 0; g < 8; ++g) {
        int p = p0 + rbase + g;
        if (p < M) xw[(long long)p * CDIM + col] = acc[g];
      }
    }
    __syncthreads();
  }
}

// ---------------- Stage 3: degrees ----------------
__global__ void deg_init_kernel(float* deg, int M) {
  int i = blockIdx.x * blockDim.x + threadIdx.x;
  if (i < M) deg[i] = 1.0f;                       // self-loop
}
__global__ void deg_acc_kernel(const int* __restrict__ ei, float* deg, int E) {
  int e = blockIdx.x * blockDim.x + threadIdx.x;
  if (e < E) atomicAdd(&deg[ei[E + e]], 1.0f);    // dst row
}
__global__ void dinv_kernel(float* deg, int M) {
  int i = blockIdx.x * blockDim.x + threadIdx.x;
  if (i < M) deg[i] = rsqrtf(deg[i]);             // deg >= 1 always
}

// ---------------- Stage 4: normalized scatter-add ----------------
__global__ void out_init_kernel(const float* __restrict__ xw, const float* __restrict__ dinv,
                                const float* __restrict__ bg, float* __restrict__ out, int M) {
  int i = blockIdx.x * blockDim.x + threadIdx.x;
  if (i < M * CDIM) {
    int m = i >> 7, c = i & 127;
    float di = dinv[m];
    out[i] = bg[c] + di * di * xw[i];             // self-loop term
  }
}
__global__ void scatter_kernel(const int* __restrict__ ei, const float* __restrict__ xw,
                               const float* __restrict__ dinv, float* __restrict__ out, int E) {
  int t = blockIdx.x * blockDim.x + threadIdx.x;
  int e = t >> 5, lane = t & 31;
  if (e < E) {
    int s = ei[e], d = ei[E + e];
    float c = dinv[s] * dinv[d];
    float4 v = reinterpret_cast<const float4*>(xw + (long long)s * CDIM)[lane];
    float* o = out + (long long)d * CDIM + lane * 4;
    atomicAdd(o + 0, c * v.x);
    atomicAdd(o + 1, c * v.y);
    atomicAdd(o + 2, c * v.z);
    atomicAdd(o + 3, c * v.w);
  }
}

// ---------------- Stage 5: pos gather ----------------
__global__ void pos_kernel(const float* __restrict__ pos, const int* __restrict__ sidx,
                           float* __restrict__ pout, int M) {
  int i = blockIdx.x * blockDim.x + threadIdx.x;
  if (i < 3 * M) {
    int p = i / 3, c = i - 3 * p;
    pout[i] = pos[(long long)sidx[p] * 3 + c];
  }
}

extern "C" void kernel_launch(void* const* d_in, const int* in_sizes, int n_in,
                              void* d_out, int out_size, void* d_ws, size_t ws_size,
                              hipStream_t stream) {
  const float* x    = (const float*)d_in[0];
  const float* pos  = (const float*)d_in[1];
  const float* W1   = (const float*)d_in[2];
  const float* b1   = (const float*)d_in[3];
  const float* W2   = (const float*)d_in[4];
  const float* b2   = (const float*)d_in[5];
  const float* Wg   = (const float*)d_in[6];
  const float* bg   = (const float*)d_in[7];
  const int*   sidx = (const int*)d_in[8];
  const int*   gidx = (const int*)d_in[9];
  const int*   ei   = (const int*)d_in[10];

  const int M = in_sizes[8];
  const int E = in_sizes[10] / 2;

  float* ws   = (float*)d_ws;
  float* pf   = ws;                               // [M,128]
  float* xw   = ws + (size_t)M * CDIM;            // [M,128]
  float* dinv = ws + (size_t)2 * M * CDIM;        // [M]

  float* xout = (float*)d_out;                    // [M,128]
  float* pout = xout + (size_t)M * CDIM;          // [M,3]

  pointnet_kernel<<<512, 256, 0, stream>>>(x, W1, b1, W2, b2, sidx, gidx, pf, M);

  deg_init_kernel<<<(M + 255) / 256, 256, 0, stream>>>(dinv, M);
  deg_acc_kernel<<<(E + 255) / 256, 256, 0, stream>>>(ei, dinv, E);
  dinv_kernel<<<(M + 255) / 256, 256, 0, stream>>>(dinv, M);

  gcn_gemm_kernel<<<391, 256, 0, stream>>>(pf, Wg, xw, M);

  out_init_kernel<<<(M * CDIM + 255) / 256, 256, 0, stream>>>(xw, dinv, bg, xout, M);
  scatter_kernel<<<(E * 32 + 255) / 256, 256, 0, stream>>>(ei, xw, dinv, xout, E);

  pos_kernel<<<(3 * M + 255) / 256, 256, 0, stream>>>(pos, sidx, pout, M);
}
